// FusionTextImageBlock_67662914781736
// MI455X (gfx1250) — matrix operand order repaired and checked by reading, not compiled, not run on hardware
//
#include <hip/hip_runtime.h>
#include <hip/hip_bf16.h>
#include <cstdint>

// ---------------------------------------------------------------------------
// CDNA5 (gfx1250) implementation of FusionTextImageBlock forward.
// All heavy GEMMs + attention use v_wmma_f32_16x16x32_bf16 (f32 accumulate).
// ---------------------------------------------------------------------------

typedef __bf16 v16bf __attribute__((ext_vector_type(16)));
typedef float  v8f   __attribute__((ext_vector_type(8)));

__device__ __forceinline__ float qgelu_f(float x) {
    return x / (1.0f + __expf(-1.702f * x));
}

// ---------------------------------------------------------------------------
// Generic GEMM:  C[M,N] = act(A[M,K] @ B[K,N] + bias) (+ res)
// A row-major (row stride lda), B row-major (row stride ldb, supports column
// slices via pointer offset), C row stride ldc. Tile 64x64, 8 waves, K-step 32.
// Uniform fast paths for interior tiles; global_prefetch of the next K-tile
// overlaps HBM latency with the current tile's WMMAs.
// ---------------------------------------------------------------------------
__global__ __launch_bounds__(256)
void gemm_wmma_k(const float* __restrict__ A, int lda,
                 const float* __restrict__ B, int ldb,
                 const float* __restrict__ bias,
                 const float* __restrict__ res, int ldr,
                 float* __restrict__ C, int ldc,
                 int M, int N, int K, int act)
{
    __shared__ __bf16 As[64][34];
    __shared__ __bf16 Bs[64][34];

    const int tid  = threadIdx.x;
    const int lane = tid & 31;
    const int wave = tid >> 5;
    const int rg   = wave >> 1;   // 0..3 : 16-row group
    const int cg   = wave & 1;    // 0..1 : 32-col group
    const int m0   = blockIdx.y * 64;
    const int n0   = blockIdx.x * 64;
    const int ml   = lane & 15;
    const int sel  = lane >> 4;

    const bool fullM = (m0 + 64 <= M);
    const bool fullN = (n0 + 64 <= N);

    v8f acc0 = {0.f,0.f,0.f,0.f,0.f,0.f,0.f,0.f};
    v8f acc1 = {0.f,0.f,0.f,0.f,0.f,0.f,0.f,0.f};

    for (int k0 = 0; k0 < K; k0 += 32) {
        __syncthreads();
        const bool fullK = (k0 + 32 <= K);

        // ---- A tile 64x32 -> bf16 LDS ----
        if (fullM && fullK) {
            float va[8];
            #pragma unroll
            for (int i = 0; i < 8; ++i) {
                int idx = tid + i * 256;
                int r = idx >> 5, kk = idx & 31;
                va[i] = A[(size_t)(m0 + r) * lda + (k0 + kk)];
            }
            #pragma unroll
            for (int i = 0; i < 8; ++i) {
                int idx = tid + i * 256;
                As[idx >> 5][idx & 31] = (__bf16)va[i];
            }
        } else {
            #pragma unroll
            for (int i = 0; i < 8; ++i) {
                int idx = tid + i * 256;
                int r = idx >> 5, kk = idx & 31;
                int gr = m0 + r, gk = k0 + kk;
                float v = (gr < M && gk < K) ? A[(size_t)gr * lda + gk] : 0.f;
                As[r][kk] = (__bf16)v;
            }
        }
        // ---- B tile 32x64, stored transposed: Bs[n][k] ----
        if (fullN && fullK) {
            float vb[8];
            #pragma unroll
            for (int i = 0; i < 8; ++i) {
                int idx = tid + i * 256;
                int nn = idx & 63, kk = idx >> 6;
                vb[i] = B[(size_t)(k0 + kk) * ldb + (n0 + nn)];
            }
            #pragma unroll
            for (int i = 0; i < 8; ++i) {
                int idx = tid + i * 256;
                Bs[idx & 63][idx >> 6] = (__bf16)vb[i];
            }
        } else {
            #pragma unroll
            for (int i = 0; i < 8; ++i) {
                int idx = tid + i * 256;
                int nn = idx & 63, kk = idx >> 6;
                int gn = n0 + nn, gk = k0 + kk;
                float v = (gn < N && gk < K) ? B[(size_t)gk * ldb + gn] : 0.f;
                Bs[nn][kk] = (__bf16)v;
            }
        }

        // ---- prefetch next K-tile (global_prefetch_b8, no counters/waits) ----
        if (k0 + 32 < K) {
            int kn = k0 + 32;
            if (tid < 64) {                     // next A tile: 64 rows x 128B
                int gr = m0 + tid;
                if (gr < M) __builtin_prefetch(&A[(size_t)gr * lda + kn], 0, 3);
            } else if (tid < 96) {              // next B tile: 32 rows, first line
                int gk = kn + (tid - 64);
                if (gk < K) __builtin_prefetch(&B[(size_t)gk * ldb + n0], 0, 3);
            } else if (tid < 128) {             // next B tile: 32 rows, second line
                int gk = kn + (tid - 96);
                if (gk < K) __builtin_prefetch(&B[(size_t)gk * ldb + n0 + 32], 0, 3);
            }
        }
        __syncthreads();

        // A fragment (rows rg*16..rg*16+15)
        v16bf a;
        #pragma unroll
        for (int j = 0; j < 16; ++j) {
            int kk = j + sel * 16;
            a[j] = As[rg * 16 + ml][kk];
        }
        // two B fragments (16 cols each)
        v16bf b0, b1;
        #pragma unroll
        for (int j = 0; j < 16; ++j) {
            int kk = j + sel * 16;
            b0[j] = Bs[cg * 32 + ml][kk];
            b1[j] = Bs[cg * 32 + 16 + ml][kk];
        }
        acc0 = __builtin_amdgcn_wmma_f32_16x16x32_bf16(false, a, false, b0, (short)0, acc0, false, false);
        acc1 = __builtin_amdgcn_wmma_f32_16x16x32_bf16(false, a, false, b1, (short)0, acc1, false, false);
    }

    // ---- epilogue ----
    if (fullM && fullN) {
        #pragma unroll
        for (int t = 0; t < 2; ++t) {
            v8f accv = t ? acc1 : acc0;
            #pragma unroll
            for (int r = 0; r < 8; ++r) {
                int row = m0 + rg * 16 + r + sel * 8;
                int col = n0 + cg * 32 + t * 16 + ml;
                float v = accv[r];
                if (bias) v += bias[col];
                if (act == 1) v = qgelu_f(v);
                if (res) v += res[(size_t)row * ldr + col];
                C[(size_t)row * ldc + col] = v;
            }
        }
    } else {
        #pragma unroll
        for (int t = 0; t < 2; ++t) {
            v8f accv = t ? acc1 : acc0;
            #pragma unroll
            for (int r = 0; r < 8; ++r) {
                int row = m0 + rg * 16 + r + sel * 8;
                int col = n0 + cg * 32 + t * 16 + ml;
                if (row < M && col < N) {
                    float v = accv[r];
                    if (bias) v += bias[col];
                    if (act == 1) v = qgelu_f(v);
                    if (res) v += res[(size_t)row * ldr + col];
                    C[(size_t)row * ldc + col] = v;
                }
            }
        }
    }
}

// ---------------------------------------------------------------------------
// Fused attention for one (n, h): scores = (Q Kᵀ)/8, softmax in LDS, O = P V.
// hd fixed at 64. One wave per block; block handles 16 query rows.
// q/k/v element (l, n, h, dd) at ptr + l*s?l + n*s?n + h*64 + dd.
// o written to (L, N, dmodel) contiguous.
// ---------------------------------------------------------------------------
__global__ __launch_bounds__(32)
void attn_fused_k(const float* __restrict__ q, long sQl, long sQn,
                  const float* __restrict__ k, long sKl, long sKn,
                  const float* __restrict__ v, long sVl, long sVn,
                  float* __restrict__ o, int dmodel,
                  int L, int S, int N, int nh, float scale)
{
    __shared__ float sc[16][200];   // S <= 197

    const int lane = threadIdx.x;
    const int ml   = lane & 15;
    const int sel  = lane >> 4;
    const int bid  = blockIdx.x;
    const int n    = bid / nh;
    const int h    = bid % nh;
    const int lblk = blockIdx.y * 16;
    const long hOff = (long)h * 64;

    // ---- Q fragments: 2 K-steps covering hd=64 ----
    v16bf aq0, aq1;
    {
        int l = lblk + ml;
        const float* qp = q + (long)l * sQl + (long)n * sQn + hOff;
        if (lblk + 16 <= L) {   // uniform fast path
            float v0[16], v1[16];
            #pragma unroll
            for (int j = 0; j < 16; ++j) {
                int kk = j + sel * 16;
                v0[j] = qp[kk];
                v1[j] = qp[kk + 32];
            }
            #pragma unroll
            for (int j = 0; j < 16; ++j) { aq0[j] = (__bf16)v0[j]; aq1[j] = (__bf16)v1[j]; }
        } else {
            bool lok = l < L;
            #pragma unroll
            for (int j = 0; j < 16; ++j) {
                int kk = j + sel * 16;
                aq0[j] = (__bf16)(lok ? qp[kk] : 0.f);
                aq1[j] = (__bf16)(lok ? qp[kk + 32] : 0.f);
            }
        }
    }

    // ---- scores ----
    for (int s0 = 0; s0 < S; s0 += 16) {
        v8f c = {0.f,0.f,0.f,0.f,0.f,0.f,0.f,0.f};
        int scol = s0 + ml;
        const float* kp = k + (long)scol * sKl + (long)n * sKn + hOff;
        v16bf bk0, bk1;
        if (s0 + 16 <= S) {     // uniform fast path
            float v0[16], v1[16];
            #pragma unroll
            for (int j = 0; j < 16; ++j) {
                int kk = j + sel * 16;
                v0[j] = kp[kk];
                v1[j] = kp[kk + 32];
            }
            #pragma unroll
            for (int j = 0; j < 16; ++j) { bk0[j] = (__bf16)v0[j]; bk1[j] = (__bf16)v1[j]; }
        } else {
            bool sok = scol < S;
            #pragma unroll
            for (int j = 0; j < 16; ++j) {
                int kk = j + sel * 16;
                bk0[j] = (__bf16)(sok ? kp[kk] : 0.f);
                bk1[j] = (__bf16)(sok ? kp[kk + 32] : 0.f);
            }
        }
        c = __builtin_amdgcn_wmma_f32_16x16x32_bf16(false, aq0, false, bk0, (short)0, c, false, false);
        c = __builtin_amdgcn_wmma_f32_16x16x32_bf16(false, aq1, false, bk1, (short)0, c, false, false);
        #pragma unroll
        for (int r = 0; r < 8; ++r) {
            int row = r + sel * 8;
            if (scol < S) sc[row][scol] = c[r] * scale;
        }
    }
    __syncthreads();

    // ---- softmax: lanes 0..15 each own one row ----
    if (lane < 16) {
        float mx = -3.0e38f;
        for (int s = 0; s < S; ++s) mx = fmaxf(mx, sc[lane][s]);
        float sum = 0.f;
        for (int s = 0; s < S; ++s) { float e = __expf(sc[lane][s] - mx); sc[lane][s] = e; sum += e; }
        float inv = 1.f / sum;
        for (int s = 0; s < S; ++s) sc[lane][s] *= inv;
    }
    __syncthreads();

    // ---- O = P V  (64 output dims = 4 column groups) ----
    v8f co[4];
    #pragma unroll
    for (int i = 0; i < 4; ++i) co[i] = (v8f){0.f,0.f,0.f,0.f,0.f,0.f,0.f,0.f};

    for (int s0 = 0; s0 < S; s0 += 32) {
        v16bf ap;
        #pragma unroll
        for (int j = 0; j < 16; ++j) {
            int kk = j + sel * 16;
            int s = s0 + kk;
            ap[j] = (__bf16)((s < S) ? sc[ml][s] : 0.f);
        }
        if (s0 + 32 <= S) {     // uniform fast path
            #pragma unroll
            for (int cgi = 0; cgi < 4; ++cgi) {
                v16bf bv;
                int d = cgi * 16 + ml;
                float vv[16];
                #pragma unroll
                for (int j = 0; j < 16; ++j) {
                    int s = s0 + j + sel * 16;
                    vv[j] = v[(long)s * sVl + (long)n * sVn + hOff + d];
                }
                #pragma unroll
                for (int j = 0; j < 16; ++j) bv[j] = (__bf16)vv[j];
                co[cgi] = __builtin_amdgcn_wmma_f32_16x16x32_bf16(false, ap, false, bv, (short)0, co[cgi], false, false);
            }
        } else {
            #pragma unroll
            for (int cgi = 0; cgi < 4; ++cgi) {
                v16bf bv;
                int d = cgi * 16 + ml;
                #pragma unroll
                for (int j = 0; j < 16; ++j) {
                    int s = s0 + j + sel * 16;
                    bv[j] = (__bf16)((s < S) ? v[(long)s * sVl + (long)n * sVn + hOff + d] : 0.f);
                }
                co[cgi] = __builtin_amdgcn_wmma_f32_16x16x32_bf16(false, ap, false, bv, (short)0, co[cgi], false, false);
            }
        }
    }

    // ---- store ----
    if (lblk + 16 <= L) {
        #pragma unroll
        for (int cgi = 0; cgi < 4; ++cgi) {
            #pragma unroll
            for (int r = 0; r < 8; ++r) {
                int row = lblk + r + sel * 8;
                int d = cgi * 16 + ml;
                o[((long)row * N + n) * dmodel + hOff + d] = co[cgi][r];
            }
        }
    } else {
        #pragma unroll
        for (int cgi = 0; cgi < 4; ++cgi) {
            #pragma unroll
            for (int r = 0; r < 8; ++r) {
                int row = lblk + r + sel * 8;
                if (row < L) {
                    int d = cgi * 16 + ml;
                    o[((long)row * N + n) * dmodel + hOff + d] = co[cgi][r];
                }
            }
        }
    }
}

// ---------------------------------------------------------------------------
// LayerNorm over last dim (supports in-place).
// ---------------------------------------------------------------------------
__global__ __launch_bounds__(256)
void layernorm_k(const float* __restrict__ x, float* __restrict__ y,
                 const float* __restrict__ g, const float* __restrict__ b,
                 int D, int rows)
{
    int row = blockIdx.x;
    if (row >= rows) return;
    const float* xr = x + (size_t)row * D;
    float s = 0.f, s2 = 0.f;
    for (int i = threadIdx.x; i < D; i += blockDim.x) {
        float v = xr[i]; s += v; s2 += v * v;
    }
    __shared__ float bs[256], bs2[256];
    bs[threadIdx.x] = s; bs2[threadIdx.x] = s2;
    __syncthreads();
    for (int off = 128; off > 0; off >>= 1) {
        if ((int)threadIdx.x < off) { bs[threadIdx.x] += bs[threadIdx.x + off]; bs2[threadIdx.x] += bs2[threadIdx.x + off]; }
        __syncthreads();
    }
    float mean = bs[0] / D;
    float var  = bs2[0] / D - mean * mean;
    float inv  = rsqrtf(var + 1e-5f);
    for (int i = threadIdx.x; i < D; i += blockDim.x) {
        float v = xr[i];
        y[(size_t)row * D + i] = (v - mean) * inv * g[i] + b[i];
    }
}

// ---------------------------------------------------------------------------
// Deterministic segment mean (_decompose). grid=(360, 8), block=256.
// dec[t][s][c] = mean over pairs p with segment(p)==s of x_text[t][p][c].
// ---------------------------------------------------------------------------
__global__ __launch_bounds__(256)
void decompose_k(const float* __restrict__ xt, const int* __restrict__ idx,
                 float* __restrict__ dec)
{
    int s = blockIdx.x;    // 0..359 (0..114 att, 115..359 obj)
    int t = blockIdx.y;    // 0..7
    int c = threadIdx.x;   // handles c and c+256
    float a0 = 0.f, a1 = 0.f, cnt = 0.f;
    for (int p = 0; p < 2000; ++p) {
        int seg = (s < 115) ? idx[2 * p] : (idx[2 * p + 1] + 115);
        if (seg == s) {
            const float* row = xt + ((size_t)t * 2000 + p) * 512;
            a0 += row[c]; a1 += row[c + 256]; cnt += 1.f;
        }
    }
    float inv = 1.f / cnt;
    size_t base = ((size_t)t * 360 + s) * 512;
    dec[base + c]       = a0 * inv;
    dec[base + c + 256] = a1 * inv;
}

// --------------------------- elementwise helpers ---------------------------
__global__ void transpose2d_k(const float* __restrict__ in, float* __restrict__ out,
                              int R, int C, size_t n)
{
    size_t tid = (size_t)blockIdx.x * blockDim.x + threadIdx.x;
    if (tid >= n) return;
    int c = (int)(tid % C); size_t r = tid / C;
    out[(size_t)c * R + r] = in[tid];
}

// (D0,D1,D2) -> out[d2][d1][d0]
__global__ void perm210_k(const float* __restrict__ in, float* __restrict__ out,
                          int D0, int D1, int D2, size_t n)
{
    size_t tid = (size_t)blockIdx.x * blockDim.x + threadIdx.x;
    if (tid >= n) return;
    int d2 = (int)(tid % D2); size_t r = tid / D2;
    int d1 = (int)(r % D1);   int d0 = (int)(r / D1);
    out[((size_t)d2 * D1 + d1) * D0 + d0] = in[tid];
}

// ykv[l][n][c] = y2[c][l]  (197,8,768) <- (768,197)
__global__ void bcast_y_k(const float* __restrict__ y2, float* __restrict__ out, size_t n)
{
    size_t tid = (size_t)blockIdx.x * blockDim.x + threadIdx.x;
    if (tid >= n) return;
    int c = (int)(tid % 768); size_t r = tid / 768;
    int l = (int)(r / 8);
    out[tid] = y2[(size_t)c * 197 + l];
}

// out[l][n][c] = dec[l%8][n][c], l in [0,64)
__global__ void tile_dec_k(const float* __restrict__ dec, float* __restrict__ out, size_t n)
{
    size_t tid = (size_t)blockIdx.x * blockDim.x + threadIdx.x;
    if (tid >= n) return;
    size_t inner = tid % (size_t)(360 * 512);
    int l = (int)(tid / (size_t)(360 * 512));
    out[tid] = dec[(size_t)(l & 7) * (360 * 512) + inner];
}

// com[m][p][c] = xt[m][att[p]][c] * xt[m][obj[p]+115][c]
__global__ void com_gather_k(const float* __restrict__ xt, const int* __restrict__ idx,
                             float* __restrict__ com, size_t n)
{
    size_t tid = (size_t)blockIdx.x * blockDim.x + threadIdx.x;
    if (tid >= n) return;
    int c = (int)(tid % 512); size_t r = tid / 512;
    int p = (int)(r % 2000);  int m = (int)(r / 2000);
    int a  = idx[2 * p];
    int ob = idx[2 * p + 1];
    float xa = xt[((size_t)m * 360 + a) * 512 + c];
    float xo = xt[((size_t)m * 360 + 115 + ob) * 512 + c];
    com[tid] = xa * xo;
}

// ---------------------------------------------------------------------------
// Host orchestration
// ---------------------------------------------------------------------------
extern "C" void kernel_launch(void* const* d_in, const int* in_sizes, int n_in,
                              void* d_out, int out_size, void* d_ws, size_t ws_size,
                              hipStream_t stream)
{
    (void)in_sizes; (void)n_in; (void)out_size;

    const float* x_image = (const float*)d_in[0];   // (197, 8, 768)
    const float* x_text  = (const float*)d_in[1];   // (8, 2000, 512)
    const int*   idx     = (const int*)d_in[2];     // (2000, 2)
    // d_in[3] = b (scalar 8)

    auto F = [&](int i) { return (const float*)d_in[i]; };
    // params leaves in JAX pytree (sorted-key) order, starting at index 4:
    // cross_img: attn(w_in,b_in,w_out,b_out), fc(w,b), ln2(g,b), ln_x(g,b), ln_y(g,b), proj(w,b)
    const int CI = 4;
    const float *ci_win = F(CI+0), *ci_bin = F(CI+1), *ci_wout = F(CI+2), *ci_bout = F(CI+3);
    const float *ci_fcw = F(CI+4), *ci_fcb = F(CI+5);
    const float *ci_l2g = F(CI+6), *ci_l2b = F(CI+7);
    const float *ci_lxg = F(CI+8), *ci_lxb = F(CI+9);
    const float *ci_lyg = F(CI+10), *ci_lyb = F(CI+11);
    const float *ci_pw  = F(CI+12), *ci_pb  = F(CI+13);
    const int CT = 18;
    const float *ct_win = F(CT+0), *ct_bin = F(CT+1), *ct_wout = F(CT+2), *ct_bout = F(CT+3);
    const float *ct_fcw = F(CT+4), *ct_fcb = F(CT+5);
    const float *ct_l2g = F(CT+6), *ct_l2b = F(CT+7);
    const float *ct_lxg = F(CT+8), *ct_lxb = F(CT+9);
    const float *ct_lyg = F(CT+10), *ct_lyb = F(CT+11);
    const float *ct_pw  = F(CT+12), *ct_pb  = F(CT+13);
    const float *fine_w = F(32), *fine_b = F(33);
    const float *i2t1_w = F(34), *i2t1_b = F(35);
    const float *i2t2_w = F(36), *i2t2_b = F(37);
    // res blocks: attn(4), fc(2), ln1(2), ln2(2), proj(2)
    const int RI = 38;
    const float *ri_win = F(RI+0), *ri_bin = F(RI+1), *ri_wout = F(RI+2), *ri_bout = F(RI+3);
    const float *ri_fcw = F(RI+4), *ri_fcb = F(RI+5);
    const float *ri_l1g = F(RI+6), *ri_l1b = F(RI+7);
    const float *ri_l2g = F(RI+8), *ri_l2b = F(RI+9);
    const float *ri_pw  = F(RI+10), *ri_pb  = F(RI+11);
    const int RT = 50;
    const float *rt_win = F(RT+0), *rt_bin = F(RT+1), *rt_wout = F(RT+2), *rt_bout = F(RT+3);
    const float *rt_fcw = F(RT+4), *rt_fcb = F(RT+5);
    const float *rt_l1g = F(RT+6), *rt_l1b = F(RT+7);
    const float *rt_l2g = F(RT+8), *rt_l2b = F(RT+9);
    const float *rt_pw  = F(RT+10), *rt_pb  = F(RT+11);
    const float *t2i1_w = F(62), *t2i1_b = F(63);
    const float *t2i2_w = F(64), *t2i2_b = F(65);

    // ----- workspace layout (float elements) -----
    size_t off = 0;
    auto alloc = [&](size_t nf) { size_t o = off; off += nf; return o; };
    const size_t oDEC  = alloc((size_t)8 * 360 * 512);       // 1,474,560
    const size_t oY1   = alloc((size_t)2880 * 768);
    const size_t oYT   = alloc((size_t)768 * 2880);
    const size_t oY2   = alloc((size_t)768 * 197);
    const size_t oYKV  = alloc((size_t)197 * 8 * 768);
    const size_t oXIA  = alloc((size_t)197 * 8 * 768);
    const size_t oXIB  = alloc((size_t)197 * 8 * 768);
    const size_t oXN   = alloc((size_t)23040 * 512);         // LN scratch (img fits too)
    const size_t oXTA  = alloc((size_t)23040 * 512);
    const size_t oXTB  = alloc((size_t)23040 * 512);
    const size_t oZ1   = alloc((size_t)197 * 8 * 512);
    const size_t oZT   = alloc((size_t)197 * 8 * 512);
    const size_t oZ2   = alloc((size_t)512 * 8 * 2880);
    const size_t oZ2P  = alloc((size_t)512 * 8 * 2880);
    // big aliased slab: qkv | o | hid, later reused for com
    const size_t oS1   = alloc((size_t)23040 * 2048 * 2);    // 94,371,840 floats
    const size_t S1Q   = oS1;                                // up to 23040x512 (or full qkv)
    const size_t S1KV  = oS1 + (size_t)23040 * 512;          // up to 23040x1024
    const size_t S1O   = oS1 + (size_t)23040 * 1536;         // 23040x512 / 1576x768
    const size_t S1HID = oS1 + (size_t)23040 * 2048;         // 23040x2048
    const size_t S1COM = oS1;                                // 128000x512 (qkv/o/hid dead)
    if (ws_size < off * sizeof(float)) return;

    float* w = (float*)d_ws;
    float* out_img = (float*)d_out;                  // 197*8*768 floats
    float* out_txt = (float*)d_out + 1210368;        // 64*2000*512 floats

    auto gemm = [&](const float* A, int lda, const float* B, int ldb,
                    const float* bias, const float* res, int ldr,
                    float* C, int ldc, int M, int N, int K, int act) {
        dim3 g((N + 63) / 64, (M + 63) / 64);
        gemm_wmma_k<<<g, 256, 0, stream>>>(A, lda, B, ldb, bias, res, ldr, C, ldc, M, N, K, act);
    };
    auto ln = [&](const float* x, float* y, const float* g_, const float* b_, int D, int rows) {
        layernorm_k<<<rows, 256, 0, stream>>>(x, y, g_, b_, D, rows);
    };
    auto attn = [&](const float* q, long sQl, long sQn,
                    const float* k, long sKl, long sKn,
                    const float* v, long sVl, long sVn,
                    float* o, int dmodel, int L, int S, int N, int nh) {
        dim3 g((unsigned)(N * nh), (unsigned)((L + 15) / 16));
        attn_fused_k<<<g, 32, 0, stream>>>(q, sQl, sQn, k, sKl, sKn, v, sVl, sVn,
                                           o, dmodel, L, S, N, nh, 0.125f);
    };
    auto ew = [&](size_t n) { return dim3((unsigned)((n + 255) / 256)); };

    // ===== 1. decompose: dec (8,360,512) =====
    decompose_k<<<dim3(360, 8), 256, 0, stream>>>(x_text, idx, w + oDEC);

    // ===== 2. y path: dec -> (768,197) -> broadcast KV =====
    gemm(w + oDEC, 512, t2i1_w, 768, t2i1_b, nullptr, 0, w + oY1, 768, 2880, 768, 512, 0);
    { size_t n = (size_t)2880 * 768; transpose2d_k<<<ew(n), 256, 0, stream>>>(w + oY1, w + oYT, 2880, 768, n); }
    gemm(w + oYT, 2880, t2i2_w, 197, t2i2_b, nullptr, 0, w + oY2, 197, 768, 197, 2880, 0);
    { size_t n = (size_t)197 * 8 * 768; bcast_y_k<<<ew(n), 256, 0, stream>>>(w + oY2, w + oYKV, n); }

    // ===== 3. cross_img =====
    ln(w + oYKV, w + oYKV, ci_lyg, ci_lyb, 768, 1576);              // yn (in place)
    ln(x_image, w + oXN, ci_lxg, ci_lxb, 768, 1576);                // xn
    gemm(w + oXN, 768, ci_win, 2304, ci_bin, nullptr, 0, w + S1Q, 768, 1576, 768, 768, 0);          // Q
    gemm(w + oYKV, 768, ci_win + 768, 2304, ci_bin + 768, nullptr, 0, w + S1KV, 1536, 1576, 1536, 768, 0); // KV
    attn(w + S1Q, 8 * 768, 768, w + S1KV, 8 * 1536, 1536, w + S1KV + 768, 8 * 1536, 1536,
         w + S1O, 768, 197, 197, 8, 12);
    gemm(w + S1O, 768, ci_wout, 768, ci_bout, x_image, 768, w + oXIA, 768, 1576, 768, 768, 0);
    ln(w + oXIA, w + oXN, ci_l2g, ci_l2b, 768, 1576);
    gemm(w + oXN, 768, ci_fcw, 3072, ci_fcb, nullptr, 0, w + S1HID, 3072, 1576, 3072, 768, 1);
    gemm(w + S1HID, 3072, ci_pw, 768, ci_pb, w + oXIA, 768, w + oXIB, 768, 1576, 768, 3072, 0);

    // ===== 4. res_img =====
    ln(w + oXIB, w + oXN, ri_l1g, ri_l1b, 768, 1576);
    gemm(w + oXN, 768, ri_win, 2304, ri_bin, nullptr, 0, w + S1Q, 2304, 1576, 2304, 768, 0); // full QKV
    attn(w + S1Q, 8 * 2304, 2304, w + S1Q + 768, 8 * 2304, 2304, w + S1Q + 1536, 8 * 2304, 2304,
         w + S1O, 768, 197, 197, 8, 12);
    gemm(w + S1O, 768, ri_wout, 768, ri_bout, w + oXIB, 768, w + oXIA, 768, 1576, 768, 768, 0);
    ln(w + oXIA, w + oXN, ri_l2g, ri_l2b, 768, 1576);
    gemm(w + oXN, 768, ri_fcw, 3072, ri_fcb, nullptr, 0, w + S1HID, 3072, 1576, 3072, 768, 1);
    gemm(w + S1HID, 3072, ri_pw, 768, ri_pb, w + oXIA, 768, w + oXIB, 768, 1576, 768, 3072, 0);
    hipMemcpyAsync(out_img, w + oXIB, (size_t)1210368 * sizeof(float),
                   hipMemcpyDeviceToDevice, stream);

    // ===== 5. z path: x_image(orig) -> (64,360,512) =====
    gemm(x_image, 768, i2t1_w, 512, i2t1_b, nullptr, 0, w + oZ1, 512, 1576, 512, 768, 0);
    { size_t n = (size_t)197 * 8 * 512;  perm210_k<<<ew(n), 256, 0, stream>>>(w + oZ1, w + oZT, 197, 8, 512, n); }
    gemm(w + oZT, 197, i2t2_w, 2880, i2t2_b, nullptr, 0, w + oZ2, 2880, 4096, 2880, 197, 0);
    { size_t n = (size_t)512 * 8 * 2880; perm210_k<<<ew(n), 256, 0, stream>>>(w + oZ2, w + oZ2P, 512, 8, 2880, n); }

    // ===== 6. cross_txt =====
    ln(w + oZ2P, w + oZ2P, ct_lyg, ct_lyb, 512, 23040);             // yn in place
    { size_t n = (size_t)23040 * 512; tile_dec_k<<<ew(n), 256, 0, stream>>>(w + oDEC, w + oXTA, n); }
    ln(w + oXTA, w + oXN, ct_lxg, ct_lxb, 512, 23040);
    gemm(w + oXN, 512, ct_win, 1536, ct_bin, nullptr, 0, w + S1Q, 512, 23040, 512, 512, 0);          // Q
    gemm(w + oZ2P, 512, ct_win + 512, 1536, ct_bin + 512, nullptr, 0, w + S1KV, 1024, 23040, 1024, 512, 0); // KV
    attn(w + S1Q, 360 * 512, 512, w + S1KV, 360 * 1024, 1024, w + S1KV + 512, 360 * 1024, 1024,
         w + S1O, 512, 64, 64, 360, 8);
    gemm(w + S1O, 512, ct_wout, 512, ct_bout, w + oXTA, 512, w + oXTB, 512, 23040, 512, 512, 0);
    ln(w + oXTB, w + oXN, ct_l2g, ct_l2b, 512, 23040);
    gemm(w + oXN, 512, ct_fcw, 2048, ct_fcb, nullptr, 0, w + S1HID, 2048, 23040, 2048, 512, 1);
    gemm(w + S1HID, 2048, ct_pw, 512, ct_pb, w + oXTB, 512, w + oXTA, 512, 23040, 512, 2048, 0);

    // ===== 7. res_txt =====
    ln(w + oXTA, w + oXN, rt_l1g, rt_l1b, 512, 23040);
    gemm(w + oXN, 512, rt_win, 1536, rt_bin, nullptr, 0, w + S1Q, 1536, 23040, 1536, 512, 0); // full QKV
    attn(w + S1Q, 360 * 1536, 1536, w + S1Q + 512, 360 * 1536, 1536, w + S1Q + 1024, 360 * 1536, 1536,
         w + S1O, 512, 64, 64, 360, 8);
    gemm(w + S1O, 512, rt_wout, 512, rt_bout, w + oXTA, 512, w + oXTB, 512, 23040, 512, 512, 0);
    ln(w + oXTB, w + oXN, rt_l2g, rt_l2b, 512, 23040);
    gemm(w + oXN, 512, rt_fcw, 2048, rt_fcb, nullptr, 0, w + S1HID, 2048, 23040, 2048, 512, 1);
    gemm(w + S1HID, 2048, rt_pw, 512, rt_pb, w + oXTB, 512, w + oXTA, 512, 23040, 512, 2048, 0);

    // ===== 8. pair gather + fine projection -> out_txt =====
    { size_t n = (size_t)64 * 2000 * 512; com_gather_k<<<ew(n), 256, 0, stream>>>(w + oXTA, idx, w + S1COM, n); }
    gemm(w + S1COM, 512, fine_w, 512, fine_b, nullptr, 0, out_txt, 512, 128000, 512, 512, 0);
}
